// WindowSelfAttention_40123584479384
// MI455X (gfx1250) — matrix operand order repaired and checked by reading, not compiled, not run on hardware
//
#include <hip/hip_runtime.h>

typedef _Float16 v16h __attribute__((ext_vector_type(16)));
typedef _Float16 v8h  __attribute__((ext_vector_type(8)));
typedef float    v8f  __attribute__((ext_vector_type(8)));

#define EMBED   512
#define HEADS   16
#define HDIM    32
#define SHIFT_  4

#define XW_LD   520               // halves per row (512 + 8 pad)
#define QK_LD   40                // 32 + 8 pad
#define AT_LD   72                // 64 + 8 pad (attn reuses Q+K region)
#define VT_LD   72                // V^T: 32 rows x (64+8) cols
#define XW_HALVES   (64 * XW_LD)                    // 33280
#define HEAD_HALVES (64*QK_LD + 64*QK_LD + 32*VT_LD) // 2560+2560+2304 = 7424
#define SMEM_HALVES (XW_HALVES + HEADS * HEAD_HALVES) // 152064 halves = 304128 B

__device__ __forceinline__ v8f zero8() {
  v8f z = {0.f,0.f,0.f,0.f,0.f,0.f,0.f,0.f};
  return z;
}

// WMMA D = A(16x32 f16) * B(32x16 f16) + C(16x16 f32)
__device__ __forceinline__ v8f wmma16(v16h a, v16h b, v8f c) {
  return __builtin_amdgcn_wmma_f32_16x16x32_f16(false, a, false, b,
                                                (short)0, c, false, false);
}

// A fragment: A[m][k] = base[(row0+m)*ld + k0+k], documented 16-bit A layout:
// lanes 0-15: M=lane,   K = {k0..k0+7, k0+16..k0+23}
// lanes16-31: M=lane-16,K = {k0+8..k0+15, k0+24..k0+31}
__device__ __forceinline__ v16h load_a16(const _Float16* base, int ld,
                                         int row0, int k0, int lane) {
  int r = row0 + (lane & 15);
  const _Float16* p = base + r * ld + k0 + ((lane >> 4) << 3);
  v8h lo = *(const v8h*)p;
  v8h hi = *(const v8h*)(p + 16);
  v16h o;
#pragma unroll
  for (int i = 0; i < 8; ++i) { o[i] = lo[i]; o[i + 8] = hi[i]; }
  return o;
}

// B fragment with B[k][n] = Src[(n0+n)*ld + k0+k]  (transposed source view):
// lanes 0-15: col n=lane,   K = {k0..k0+7, k0+16..k0+23}
// lanes16-31: col n=lane-16,K = {k0+8..k0+15, k0+24..k0+31}
__device__ __forceinline__ v16h load_bt16(const _Float16* base, int ld,
                                          int n0, int k0, int lane) {
  int n = n0 + (lane & 15);
  const _Float16* p = base + n * ld + k0 + ((lane >> 4) << 3);
  v8h lo = *(const v8h*)p;
  v8h hi = *(const v8h*)(p + 16);
  v16h o;
#pragma unroll
  for (int i = 0; i < 8; ++i) { o[i] = lo[i]; o[i + 8] = hi[i]; }
  return o;
}

// C/D 16x16 f32 layout: VGPR i -> M=row0+i (+8 for hi half-wave), N=col0+(lane&15)
__device__ __forceinline__ void store_c16(v8f c, _Float16* base, int ld,
                                          int row0, int col0, int lane) {
  int r0 = row0 + ((lane >> 4) << 3);
  int n  = col0 + (lane & 15);
#pragma unroll
  for (int i = 0; i < 8; ++i) base[(r0 + i) * ld + n] = (_Float16)c[i];
}

__device__ __forceinline__ void store_cT16(v8f c, _Float16* base, int ld,
                                           int row0, int col0, int lane) {
  int r0 = row0 + ((lane >> 4) << 3);
  int n  = col0 + (lane & 15);
#pragma unroll
  for (int i = 0; i < 8; ++i) base[n * ld + (r0 + i)] = (_Float16)c[i];
}

__global__ __launch_bounds__(512) void convert_w(const float* __restrict__ wqkv,
                                                 const float* __restrict__ wout,
                                                 _Float16* __restrict__ wq16,
                                                 _Float16* __restrict__ wo16) {
  int i = blockIdx.x * 256 + threadIdx.x;
  const int NQ = 3 * EMBED * EMBED;   // 786432
  if (i < NQ) wq16[i] = (_Float16)wqkv[i];
  else        wo16[i - NQ] = (_Float16)wout[i - NQ];
}

__global__ __launch_bounds__(512) void swin_window_attn(
    const float*    __restrict__ x,     // [16][64][64][512] fp32
    const _Float16* __restrict__ wq16,  // [1536][512] f16
    const float*    __restrict__ bqkv,  // [1536]
    const _Float16* __restrict__ wo16,  // [512][512] f16
    const float*    __restrict__ bout,  // [512]
    float*          __restrict__ out)   // [16][64][64][512] fp32
{
  extern __shared__ _Float16 smem[];
  const int tid  = threadIdx.x;
  const int lane = tid & 31;
  const int wave = tid >> 5;             // head id, 0..15
  const int blk  = blockIdx.x;           // 0..1023
  const int bimg = blk >> 6;
  const int wy   = (blk >> 3) & 7;
  const int wx   = blk & 7;

  // ---------- Phase 1: load shifted window (roll -4,-4) into LDS as f16 ----------
  {
    const size_t imgbase = (size_t)bimg * (64 * 64 * 512);
    for (int idx = tid; idx < 64 * 128; idx += 512) {
      int m  = idx >> 7;                 // window-local token 0..63
      int c4 = (idx & 127) << 2;         // channel (float4 granularity)
      int gr = ((wy << 3) + (m >> 3) + SHIFT_) & 63;
      int gc = ((wx << 3) + (m & 7)  + SHIFT_) & 63;
      const float4 v = *(const float4*)(x + imgbase + ((size_t)gr * 64 + gc) * 512 + c4);
      _Float16* d = &smem[m * XW_LD + c4];
      d[0] = (_Float16)v.x; d[1] = (_Float16)v.y;
      d[2] = (_Float16)v.z; d[3] = (_Float16)v.w;
    }
  }
  __syncthreads();

  _Float16* Qh  = smem + XW_HALVES + wave * HEAD_HALVES;   // 64 x QK_LD
  _Float16* Kh  = Qh + 64 * QK_LD;                         // 64 x QK_LD
  _Float16* VTh = Qh + 2 * 64 * QK_LD;                     // 32 x VT_LD (V^T)

  const float scale = 0.17677669529663687f;                // 1/sqrt(32)

  // ---------- Phase 2: QKV projection, one pass each (keeps ~90 VGPRs live) ----------
#pragma unroll 1
  for (int pass = 0; pass < 3; ++pass) {
    const int wrow0 = pass * EMBED + wave * HDIM;          // rows of w_qkv
    const _Float16* wb = wq16 + (size_t)wrow0 * EMBED;
    v8f acc[4][2];
#pragma unroll
    for (int mt = 0; mt < 4; ++mt) { acc[mt][0] = zero8(); acc[mt][1] = zero8(); }

#pragma unroll 2
    for (int k0 = 0; k0 < EMBED; k0 += 32) {
      v16h b0 = load_bt16(wb, EMBED, 0,  k0, lane);        // B[k][n]=W[n][k]
      v16h b1 = load_bt16(wb, EMBED, 16, k0, lane);
#pragma unroll
      for (int mt = 0; mt < 4; ++mt) {
        v16h a = load_a16(smem, XW_LD, mt * 16, k0, lane);
        acc[mt][0] = wmma16(a, b0, acc[mt][0]);
        acc[mt][1] = wmma16(a, b1, acc[mt][1]);
      }
    }
    const float smul = (pass == 0) ? scale : 1.0f;
#pragma unroll
    for (int mt = 0; mt < 4; ++mt)
#pragma unroll
      for (int nt = 0; nt < 2; ++nt) {
        float bias = bqkv[wrow0 + nt * 16 + (lane & 15)];
        v8f c = acc[mt][nt];
#pragma unroll
        for (int i = 0; i < 8; ++i) c[i] = (c[i] + bias) * smul;
        if (pass == 0)      store_c16 (c, Qh,  QK_LD, mt * 16, nt * 16, lane);
        else if (pass == 1) store_c16 (c, Kh,  QK_LD, mt * 16, nt * 16, lane);
        else                store_cT16(c, VTh, VT_LD, mt * 16, nt * 16, lane);
      }
  }
  __syncthreads();    // everyone is done reading xw; region becomes outbuf

  // ---------- Phase 3: scores = Q K^T (d=32 == one K-chunk), mask, softmax ----------
  v16h aq[4], bk[4];
#pragma unroll
  for (int mt = 0; mt < 4; ++mt) aq[mt] = load_a16(Qh, QK_LD, mt * 16, 0, lane);
#pragma unroll
  for (int nt = 0; nt < 4; ++nt) bk[nt] = load_bt16(Kh, QK_LD, nt * 16, 0, lane);

  // key-group ids for Swin mask (column = key index per lane)
  int kg[4];
#pragma unroll
  for (int nt = 0; nt < 4; ++nt) {
    int kn = nt * 16 + (lane & 15);
    kg[nt] = ((wy == 7 && (kn >> 3) >= 4) ? 2 : 0) |
             ((wx == 7 && (kn & 7)  >= 4) ? 1 : 0);
  }
  const bool masked_win = (wy == 7) || (wx == 7);

  _Float16* An = Qh;                      // attn (64 x AT_LD) reuses Q+K region

#pragma unroll 1
  for (int mt = 0; mt < 4; ++mt) {
    v8f sc[4];
#pragma unroll
    for (int nt = 0; nt < 4; ++nt) sc[nt] = wmma16(aq[mt], bk[nt], zero8());

    if (masked_win) {
#pragma unroll
      for (int i = 0; i < 8; ++i) {
        int qm = mt * 16 + ((lane >> 4) << 3) + i;
        int qg = ((wy == 7 && (qm >> 3) >= 4) ? 2 : 0) |
                 ((wx == 7 && (qm & 7)  >= 4) ? 1 : 0);
#pragma unroll
        for (int nt = 0; nt < 4; ++nt)
          if (qg != kg[nt]) sc[nt][i] = -1e9f;
      }
    }
    // row softmax: row lives across 16 lanes (same half) x 4 nt-tiles
#pragma unroll
    for (int i = 0; i < 8; ++i) {
      float m = fmaxf(fmaxf(sc[0][i], sc[1][i]), fmaxf(sc[2][i], sc[3][i]));
      m = fmaxf(m, __shfl_xor(m, 8, 32));
      m = fmaxf(m, __shfl_xor(m, 4, 32));
      m = fmaxf(m, __shfl_xor(m, 2, 32));
      m = fmaxf(m, __shfl_xor(m, 1, 32));
      float s = 0.f;
#pragma unroll
      for (int nt = 0; nt < 4; ++nt) {
        float e = __expf(sc[nt][i] - m);
        sc[nt][i] = e; s += e;
      }
      s += __shfl_xor(s, 8, 32);
      s += __shfl_xor(s, 4, 32);
      s += __shfl_xor(s, 2, 32);
      s += __shfl_xor(s, 1, 32);
      float inv = 1.0f / s;
#pragma unroll
      for (int nt = 0; nt < 4; ++nt) sc[nt][i] *= inv;
    }
#pragma unroll
    for (int nt = 0; nt < 4; ++nt)
      store_c16(sc[nt], An, AT_LD, mt * 16, nt * 16, lane);
  }

  // ---------- Phase 4: out_h = attn @ V  (B[k][n] = V[k][n] = VT[n][k]) ----------
  v8f oa[4][2];
#pragma unroll
  for (int mt = 0; mt < 4; ++mt) { oa[mt][0] = zero8(); oa[mt][1] = zero8(); }
#pragma unroll
  for (int kt = 0; kt < 2; ++kt) {
    v16h b0 = load_bt16(VTh, VT_LD, 0,  kt * 32, lane);
    v16h b1 = load_bt16(VTh, VT_LD, 16, kt * 32, lane);
#pragma unroll
    for (int mt = 0; mt < 4; ++mt) {
      v16h a = load_a16(An, AT_LD, mt * 16, kt * 32, lane);
      oa[mt][0] = wmma16(a, b0, oa[mt][0]);
      oa[mt][1] = wmma16(a, b1, oa[mt][1]);
    }
  }
  // stage head output (f16) into shared outbuf = old xw region
#pragma unroll
  for (int mt = 0; mt < 4; ++mt)
#pragma unroll
    for (int nt = 0; nt < 2; ++nt)
      store_c16(oa[mt][nt], smem, XW_LD, mt * 16, wave * HDIM + nt * 16, lane);
  __syncthreads();

  // ---------- Phase 5: out = concat(out_h) @ w_out^T + b_out, inverse roll ----------
  v8f fo[4][2];
#pragma unroll
  for (int mt = 0; mt < 4; ++mt) { fo[mt][0] = zero8(); fo[mt][1] = zero8(); }
  const _Float16* wob = wo16 + (size_t)(wave * HDIM) * EMBED;
#pragma unroll 2
  for (int k0 = 0; k0 < EMBED; k0 += 32) {
    v16h b0 = load_bt16(wob, EMBED, 0,  k0, lane);
    v16h b1 = load_bt16(wob, EMBED, 16, k0, lane);
#pragma unroll
    for (int mt = 0; mt < 4; ++mt) {
      v16h a = load_a16(smem, XW_LD, mt * 16, k0, lane);
      fo[mt][0] = wmma16(a, b0, fo[mt][0]);
      fo[mt][1] = wmma16(a, b1, fo[mt][1]);
    }
  }
  const size_t obase = (size_t)bimg * (64 * 64 * 512);
#pragma unroll
  for (int mt = 0; mt < 4; ++mt)
#pragma unroll
    for (int nt = 0; nt < 2; ++nt) {
      int n = wave * HDIM + nt * 16 + (lane & 15);
      float bias = bout[n];
      int r0 = mt * 16 + ((lane >> 4) << 3);
#pragma unroll
      for (int i = 0; i < 8; ++i) {
        int m  = r0 + i;
        int gr = ((wy << 3) + (m >> 3) + SHIFT_) & 63;   // roll(+4) == inverse of load
        int gc = ((wx << 3) + (m & 7)  + SHIFT_) & 63;
        out[obase + ((size_t)gr * 64 + gc) * 512 + n] = fo[mt][nt][i] + bias;
      }
    }
}

extern "C" void kernel_launch(void* const* d_in, const int* in_sizes, int n_in,
                              void* d_out, int out_size, void* d_ws, size_t ws_size,
                              hipStream_t stream) {
  (void)in_sizes; (void)n_in; (void)out_size; (void)ws_size;
  const float* x    = (const float*)d_in[0];
  const float* wqkv = (const float*)d_in[1];
  const float* bqkv = (const float*)d_in[2];
  const float* wout = (const float*)d_in[3];
  const float* bout = (const float*)d_in[4];
  float* out = (float*)d_out;

  _Float16* wq16 = (_Float16*)d_ws;                  // 1536*512 f16
  _Float16* wo16 = wq16 + 3 * EMBED * EMBED;         // 512*512 f16

  convert_w<<<4096, 256, 0, stream>>>(wqkv, wout, wq16, wo16);

  const size_t shmem = (size_t)SMEM_HALVES * sizeof(_Float16);   // 304128 B
  hipFuncSetAttribute((const void*)swin_window_attn,
                      hipFuncAttributeMaxDynamicSharedMemorySize, (int)shmem);
  swin_window_attn<<<1024, 512, shmem, stream>>>(x, wq16, bqkv, wo16, bout, out);
}
